// BDH_1726576853700
// MI455X (gfx1250) — matrix operand order repaired and checked by reading, not compiled, not run on hardware
//
#include <hip/hip_runtime.h>

// ---------------- model constants (all compile-time) ----------------
constexpr int T_  = 1024;   // sequence length
constexpr int D_  = 256;    // model dim
constexpr int NH_ = 4;      // heads
constexpr int N_  = 8192;   // sparse dim per head
constexpr int V_  = 256;    // vocab
constexpr float EPS_ = 1e-5f;
constexpr float TWO_PI_ = 6.283185307179586f;

typedef __bf16 bf16;
typedef __bf16 v16bf __attribute__((ext_vector_type(16)));
typedef float  v8f   __attribute__((ext_vector_type(8)));

__device__ __forceinline__ bf16  f2b(float f) { return (bf16)f; }
__device__ __forceinline__ float b2f(bf16 b)  { return (float)b; }

// ---------------- WMMA GEMM: Out = epilogue(A[M,K] * Bt[NN,K]^T) ----------------
// A row-major bf16 (lda = K, or head-blocked), Bt row-major bf16 [NN][K].
// Tiles staged via async global->LDS copies (ASYNCcnt), double-buffered LDS,
// one barrier per k-tile; copy of tile k+1 overlaps WMMA compute of tile k.
constexpr int BM = 128, BN = 128, BK = 64, LDP = 72; // LDS padded stride (bf16)
enum { EPI_F32 = 0, EPI_RELU_BF = 1, EPI_CAUSAL_BF = 2, EPI_RELUMUL_BF = 3 };

template <int M, int NN, int K, int EPI, bool HB>
__global__ __launch_bounds__(256) void gemm_k(const bf16* __restrict__ A,
                                              const bf16* __restrict__ Bt,
                                              void* __restrict__ OutV,
                                              const bf16* __restrict__ Mul,
                                              long long AZ, long long BZ, long long OZ) {
  __shared__ __align__(16) bf16 Al[2][BM * LDP];
  __shared__ __align__(16) bf16 Bl[2][BN * LDP];

  const int tid  = threadIdx.x;
  const int lane = tid & 31;
  const int w    = tid >> 5;
  const int l16  = lane & 15;
  const int hf   = lane >> 4;
  const int z    = blockIdx.z;
  const int blockRow = blockIdx.y * BM;
  const int blockCol = blockIdx.x * BN;
  const int waveRow  = (w >> 2) * 64;   // 2 waves in M
  const int waveCol  = (w & 3) * 32;    // 4 waves in N

  A  += (long long)z * AZ;
  Bt += (long long)z * BZ;

  // Strict lower-triangular: blocks entirely at/above diagonal are all zero.
  if (EPI == EPI_CAUSAL_BF && blockCol >= blockRow + BM) {
    bf16* o = (bf16*)OutV + (long long)z * OZ;
#pragma unroll
    for (int mi = 0; mi < 4; ++mi)
#pragma unroll
      for (int ni = 0; ni < 2; ++ni)
#pragma unroll
        for (int v = 0; v < 8; ++v) {
          int row = blockRow + waveRow + mi * 16 + v + 8 * hf;
          int col = blockCol + waveCol + ni * 16 + l16;
          o[(long long)row * NN + col] = f2b(0.0f);
        }
    return;
  }

  v8f acc[4][2];
  const v8f zero8 = {0.f, 0.f, 0.f, 0.f, 0.f, 0.f, 0.f, 0.f};
#pragma unroll
  for (int mi = 0; mi < 4; ++mi)
#pragma unroll
    for (int ni = 0; ni < 2; ++ni) acc[mi][ni] = zero8;

  constexpr int KT = K / BK;
  constexpr long long HS = (long long)T_ * N_;  // head stride for head-blocked A

  // Issue async global->LDS copies for one 128x64 A tile + 128x64 Bt tile.
  // Each of the 256 threads issues 4+4 x 16B lane-copies.
  auto issue_tile = [&](int k0, int buf) {
    const bf16* aptr;
    int lda;
    if (HB) { lda = N_; aptr = A + (long long)(k0 >> 13) * HS + (k0 & (N_ - 1)); }
    else    { lda = K;  aptr = A + k0; }
    const bf16* bptr = Bt + k0;
#pragma unroll
    for (int r = 0; r < 4; ++r) {
      int e = r * 2048 + tid * 8;
      int row = e >> 6, col = e & 63;
      unsigned la = (unsigned)(size_t)(const void*)&Al[buf][row * LDP + col];
      unsigned long long ga =
          (unsigned long long)(const void*)(aptr + (long long)(blockRow + row) * lda + col);
      asm volatile("global_load_async_to_lds_b128 %0, %1, off"
                   :: "v"(la), "v"(ga) : "memory");
      unsigned lb = (unsigned)(size_t)(const void*)&Bl[buf][row * LDP + col];
      unsigned long long gb =
          (unsigned long long)(const void*)(bptr + (long long)(blockCol + row) * (long long)K + col);
      asm volatile("global_load_async_to_lds_b128 %0, %1, off"
                   :: "v"(lb), "v"(gb) : "memory");
    }
  };

  issue_tile(0, 0);

#pragma unroll 1
  for (int kt = 0; kt < KT; ++kt) {
    // my async copies for tile kt complete -> barrier makes it true for all waves
    asm volatile("s_wait_asynccnt 0" ::: "memory");
    __syncthreads();
    // kick off tile kt+1 into the other buffer (overlaps with compute below);
    // safe: all waves finished reading buf[(kt+1)&1] (their compute of kt-1)
    if (kt + 1 < KT) issue_tile((kt + 1) * BK, (kt + 1) & 1);

    const bf16* Ab = &Al[kt & 1][0];
    const bf16* Bb = &Bl[kt & 1][0];

#pragma unroll
    for (int ks = 0; ks < 2; ++ks) {
      union Frag { v16bf v; unsigned u[8]; };
      Frag af[4], bfm[2];
      // A fragment layout (16-bit A 16x32): k = 16*(v>=4) + 8*half + 2*(v&3)
#pragma unroll
      for (int mi = 0; mi < 4; ++mi) {
        int m = waveRow + mi * 16 + l16;
#pragma unroll
        for (int vv = 0; vv < 8; ++vv) {
          int k = ks * 32 + ((vv & 4) << 2) + hf * 8 + ((vv & 3) << 1);
          af[mi].u[vv] = *(const unsigned*)&Ab[m * LDP + k];
        }
      }
      // B fragment layout (16-bit B 32x16): k = 16*(v>=4) + 8*half + 2*(v&3)
#pragma unroll
      for (int ni = 0; ni < 2; ++ni) {
        int n = waveCol + ni * 16 + l16;
#pragma unroll
        for (int vv = 0; vv < 8; ++vv) {
          int k = ks * 32 + ((vv & 4) << 2) + hf * 8 + ((vv & 3) << 1);
          bfm[ni].u[vv] = *(const unsigned*)&Bb[n * LDP + k];
        }
      }
#pragma unroll
      for (int mi = 0; mi < 4; ++mi)
#pragma unroll
        for (int ni = 0; ni < 2; ++ni)
          acc[mi][ni] = __builtin_amdgcn_wmma_f32_16x16x32_bf16(
              false, af[mi].v, false, bfm[ni].v, (short)0, acc[mi][ni], false, false);
    }
  }

  // Epilogue: D layout => m = v + 8*half, n = lane%16
#pragma unroll
  for (int mi = 0; mi < 4; ++mi)
#pragma unroll
    for (int ni = 0; ni < 2; ++ni)
#pragma unroll
      for (int v = 0; v < 8; ++v) {
        int row = blockRow + waveRow + mi * 16 + v + 8 * hf;
        int col = blockCol + waveCol + ni * 16 + l16;
        float val = acc[mi][ni][v];
        long long oi = (long long)row * NN + col;
        if (EPI == EPI_F32) {
          ((float*)OutV + (long long)z * OZ)[oi] = val;
        } else if (EPI == EPI_RELU_BF) {
          ((bf16*)OutV + (long long)z * OZ)[oi] = f2b(fmaxf(val, 0.f));
        } else if (EPI == EPI_CAUSAL_BF) {
          ((bf16*)OutV + (long long)z * OZ)[oi] = (col < row) ? f2b(val) : f2b(0.f);
        } else {  // EPI_RELUMUL_BF: xy = relu(acc) * x_sparse
          float mv = b2f((Mul + (long long)z * OZ)[oi]);
          ((bf16*)OutV + (long long)z * OZ)[oi] = f2b(fmaxf(val, 0.f) * mv);
        }
      }
}

// ---------------- elementwise / LN kernels ----------------
__global__ __launch_bounds__(256) void embed_ln_k(const int* __restrict__ idx,
                                                  const float* __restrict__ emb,
                                                  float* __restrict__ x,
                                                  bf16* __restrict__ xb,
                                                  bf16* __restrict__ xT) {
  __shared__ float sA[256], sB[256];
  int t = blockIdx.x, d = threadIdx.x;
  float v = emb[(long long)idx[t] * D_ + d];
  sA[d] = v; sB[d] = v * v; __syncthreads();
  for (int s = 128; s > 0; s >>= 1) {
    if (d < s) { sA[d] += sA[d + s]; sB[d] += sB[d + s]; }
    __syncthreads();
  }
  float mu = sA[0] * (1.f / D_);
  float var = sB[0] * (1.f / D_) - mu * mu;
  float y = (v - mu) * rsqrtf(var + EPS_);
  x[t * D_ + d] = y;
  xb[t * D_ + d] = f2b(y);
  xT[(long long)d * T_ + t] = f2b(y);
}

__global__ __launch_bounds__(256) void ln_rows_k(const float* __restrict__ in,
                                                 bf16* __restrict__ out) {
  __shared__ float sA[256], sB[256];
  int r = blockIdx.x, d = threadIdx.x;
  float v = in[(long long)r * D_ + d];
  sA[d] = v; sB[d] = v * v; __syncthreads();
  for (int s = 128; s > 0; s >>= 1) {
    if (d < s) { sA[d] += sA[d + s]; sB[d] += sB[d + s]; }
    __syncthreads();
  }
  float mu = sA[0] * (1.f / D_);
  float var = sB[0] * (1.f / D_) - mu * mu;
  out[(long long)r * D_ + d] = f2b((v - mu) * rsqrtf(var + EPS_));
}

// x = LN(x + LN(ymlp)); refresh f32/bf16/bf16-transposed copies of x
__global__ __launch_bounds__(256) void resid_ln_k(const float* __restrict__ ymlp,
                                                  float* __restrict__ x,
                                                  bf16* __restrict__ xb,
                                                  bf16* __restrict__ xT) {
  __shared__ float sA[256], sB[256];
  int t = blockIdx.x, d = threadIdx.x;
  float ym = ymlp[t * D_ + d];
  sA[d] = ym; sB[d] = ym * ym; __syncthreads();
  for (int s = 128; s > 0; s >>= 1) {
    if (d < s) { sA[d] += sA[d + s]; sB[d] += sB[d + s]; }
    __syncthreads();
  }
  float mu = sA[0] * (1.f / D_);
  float var = sB[0] * (1.f / D_) - mu * mu;
  float y1 = (ym - mu) * rsqrtf(var + EPS_);
  float v = x[t * D_ + d] + y1;
  __syncthreads();
  sA[d] = v; sB[d] = v * v; __syncthreads();
  for (int s = 128; s > 0; s >>= 1) {
    if (d < s) { sA[d] += sA[d + s]; sB[d] += sB[d + s]; }
    __syncthreads();
  }
  mu = sA[0] * (1.f / D_);
  var = sB[0] * (1.f / D_) - mu * mu;
  float y2 = (v - mu) * rsqrtf(var + EPS_);
  x[t * D_ + d] = y2;
  xb[t * D_ + d] = f2b(y2);
  xT[(long long)d * T_ + t] = f2b(y2);
}

// RoPE: qr = rope(phases, x_sparse); pair j shares freq 2^(-j/256)/(2*pi)
__global__ __launch_bounds__(256) void rope_k(const bf16* __restrict__ xs,
                                              bf16* __restrict__ qr) {
  long long i = (long long)blockIdx.x * 256 + threadIdx.x;  // pair index
  int j = (int)(i & (N_ / 2 - 1));
  long long ht = i >> 12;  // (h*T + t)
  int t = (int)(ht & (T_ - 1));
  long long e0 = ht * N_ + 2 * (long long)j;
  float freq = exp2f(-(float)j * (1.0f / 256.0f)) * (1.0f / TWO_PI_);
  float ph = (float)t * freq;
  ph = (ph - floorf(ph)) * TWO_PI_;
  float s = __sinf(ph), c = __cosf(ph);
  float v0 = b2f(xs[e0]), v1 = b2f(xs[e0 + 1]);
  qr[e0]     = f2b(v0 * c - v1 * s);
  qr[e0 + 1] = f2b(v1 * c + v0 * s);
}

// encoder / encoder_v: [NH][D][N] f32 -> [NH][N][D] bf16 (n-major for Bt staging)
__global__ __launch_bounds__(256) void cvt_enc_k(const float* __restrict__ in,
                                                 bf16* __restrict__ out) {
  long long i = (long long)blockIdx.x * 256 + threadIdx.x;  // < NH*D*N
  int n = (int)(i & (N_ - 1));
  long long hd = i >> 13;
  int d = (int)(hd & (D_ - 1));
  int h = (int)(hd >> 8);
  out[((long long)h * N_ + n) * D_ + d] = f2b(in[i]);
}

// generic [R][C] f32 -> [C][R] bf16
__global__ __launch_bounds__(256) void cvt_t2d_k(const float* __restrict__ in,
                                                 bf16* __restrict__ out, int R, int C) {
  long long i = (long long)blockIdx.x * 256 + threadIdx.x;  // < R*C
  int c = (int)(i % C);
  long long r = i / C;
  out[(long long)c * R + r] = f2b(in[i]);
}

// ---------------- host orchestration ----------------
extern "C" void kernel_launch(void* const* d_in, const int* in_sizes, int n_in,
                              void* d_out, int out_size, void* d_ws, size_t ws_size,
                              hipStream_t stream) {
  (void)in_sizes; (void)n_in; (void)out_size; (void)ws_size;
  const int*   idx   = (const int*)d_in[0];
  const float* embed = (const float*)d_in[1];
  const float* enc   = (const float*)d_in[2];
  const float* encv  = (const float*)d_in[3];
  const float* dec   = (const float*)d_in[4];
  const float* lmh   = (const float*)d_in[5];
  float* out = (float*)d_out;

  char* p = (char*)d_ws;
  auto carve = [&](size_t bytes) -> char* {
    char* r = p;
    p += (bytes + 255) & ~(size_t)255;
    return r;
  };
  bf16* enc_t  = (bf16*)carve((size_t)NH_ * N_ * D_ * 2);  // [NH][N][D]
  bf16* encv_t = (bf16*)carve((size_t)NH_ * N_ * D_ * 2);  // [NH][N][D]
  bf16* dec_t  = (bf16*)carve((size_t)NH_ * N_ * D_ * 2);  // [D][NH*N]
  bf16* lm_t   = (bf16*)carve((size_t)V_ * D_ * 2);        // [V][D]
  float* x_f   = (float*)carve((size_t)T_ * D_ * 4);
  bf16* x_b    = (bf16*)carve((size_t)T_ * D_ * 2);
  bf16* xT_b   = (bf16*)carve((size_t)D_ * T_ * 2);
  bf16* xs_b   = (bf16*)carve((size_t)NH_ * T_ * N_ * 2);  // x_sparse
  bf16* qr_b   = (bf16*)carve((size_t)NH_ * T_ * N_ * 2);  // qr, reused as xy
  bf16* sc_b   = (bf16*)carve((size_t)NH_ * T_ * T_ * 2);  // scores
  float* ykv_f = (float*)carve((size_t)NH_ * T_ * D_ * 4);
  bf16* ykv_b  = (bf16*)carve((size_t)NH_ * T_ * D_ * 2);
  float* ymlp_f = (float*)carve((size_t)T_ * D_ * 4);

  dim3 blk(256);

  // weight conversion (bf16, Bt-friendly layouts) — redone every call (stateless)
  cvt_enc_k<<<(NH_ * D_ * N_) / 256, blk, 0, stream>>>(enc, enc_t);
  cvt_enc_k<<<(NH_ * D_ * N_) / 256, blk, 0, stream>>>(encv, encv_t);
  cvt_t2d_k<<<(NH_ * N_ * D_) / 256, blk, 0, stream>>>(dec, dec_t, NH_ * N_, D_);
  cvt_t2d_k<<<(D_ * V_) / 256, blk, 0, stream>>>(lmh, lm_t, D_, V_);

  // x = LN(embed[idx])
  embed_ln_k<<<T_, blk, 0, stream>>>(idx, embed, x_f, x_b, xT_b);

  for (int layer = 0; layer < 3; ++layer) {
    // x_sparse = relu(x @ encoder[h])            [NH,T,N]
    gemm_k<T_, N_, D_, EPI_RELU_BF, false>
        <<<dim3(N_ / BN, T_ / BM, NH_), blk, 0, stream>>>(
            x_b, enc_t, xs_b, nullptr, 0, (long long)N_ * D_, (long long)T_ * N_);
    // qr = rope(x_sparse)
    rope_k<<<(NH_ * T_ * (N_ / 2)) / 256, blk, 0, stream>>>(xs_b, qr_b);
    // scores = tril(qr @ qr^T, -1)               [NH,T,T]
    gemm_k<T_, T_, N_, EPI_CAUSAL_BF, false>
        <<<dim3(T_ / BN, T_ / BM, NH_), blk, 0, stream>>>(
            qr_b, qr_b, sc_b, nullptr, (long long)T_ * N_, (long long)T_ * N_,
            (long long)T_ * T_);
    // ykv = scores @ x                           [NH,T,D]
    gemm_k<T_, D_, T_, EPI_F32, false>
        <<<dim3(D_ / BN, T_ / BM, NH_), blk, 0, stream>>>(
            sc_b, xT_b, ykv_f, nullptr, (long long)T_ * T_, 0, (long long)T_ * D_);
    // ykv = LN(ykv)
    ln_rows_k<<<NH_ * T_, blk, 0, stream>>>(ykv_f, ykv_b);
    // xy = relu(ykv @ encoder_v[h]) * x_sparse   [NH,T,N] (reuses qr buffer)
    gemm_k<T_, N_, D_, EPI_RELUMUL_BF, false>
        <<<dim3(N_ / BN, T_ / BM, NH_), blk, 0, stream>>>(
            ykv_b, encv_t, qr_b, xs_b, (long long)T_ * D_, (long long)N_ * D_,
            (long long)T_ * N_);
    // ymlp = xy_flat @ decoder  (head-blocked A, K = NH*N = 32768)
    gemm_k<T_, D_, NH_ * N_, EPI_F32, true>
        <<<dim3(D_ / BN, T_ / BM, 1), blk, 0, stream>>>(
            qr_b, dec_t, ymlp_f, nullptr, 0, 0, 0);
    // x = LN(x + LN(ymlp))
    resid_ln_k<<<T_, blk, 0, stream>>>(ymlp_f, x_f, x_b, xT_b);
  }

  // logits = x @ lm_head
  gemm_k<T_, V_, D_, EPI_F32, false>
      <<<dim3(V_ / BN, T_ / BM, 1), blk, 0, stream>>>(x_b, lm_t, out, nullptr, 0, 0, 0);
}
